// MainModel_35081292873855
// MI455X (gfx1250) — matrix operand order repaired and checked by reading, not compiled, not run on hardware
//
#include <hip/hip_runtime.h>

#define BB   4
#define NN   1024
#define DD   256
#define HH   4
#define DIMH 64
#define LL   18

typedef __attribute__((ext_vector_type(16))) _Float16 v16h;
typedef __attribute__((ext_vector_type(8)))  _Float16 v8h;
typedef __attribute__((ext_vector_type(8)))  float    v8f;

// ---------------------------------------------------------------------------
// gfx1250 WMMA 16x16x32 f16 fragment layouts (CDNA5 ISA 7.12.2):
// A (16x32): lane&15 = row M; per lane two contiguous 8-half runs at
//            K = 8*hi .. 8*hi+7 and 16+8*hi .. 23+8*hi   (hi = lane>=16)
// B (32x16): lane&15 = col N; per lane two contiguous 8-half runs at
//            K = 8*hi .. 8*hi+7 and 16+8*hi .. 23+8*hi   (same as A)
// C/D (16x16 f32): VGPR r holds M = r + 8*hi, N = lane&15.
// With f16 data stored K-contiguous, fragments are pure b128 loads.
// ---------------------------------------------------------------------------
__device__ __forceinline__ v16h ld_a_frag(const _Float16* rowK, int hi) {
  v8h p0 = *(const v8h*)(rowK + 8 * hi);
  v8h p1 = *(const v8h*)(rowK + 16 + 8 * hi);
  return __builtin_shufflevector(p0, p1, 0, 1, 2, 3, 4, 5, 6, 7, 8, 9, 10, 11,
                                 12, 13, 14, 15);
}
__device__ __forceinline__ v16h ld_b_frag(const _Float16* rowK, int hi) {
  v8h p0 = *(const v8h*)(rowK + 8 * hi);
  v8h p1 = *(const v8h*)(rowK + 16 + 8 * hi);
  return __builtin_shufflevector(p0, p1, 0, 1, 2, 3, 4, 5, 6, 7, 8, 9, 10, 11,
                                 12, 13, 14, 15);
}
__device__ __forceinline__ v8f wmma16(v16h a, v16h b, v8f c) {
  return __builtin_amdgcn_wmma_f32_16x16x32_f16(false, a, false, b, (short)0, c,
                                                false, false);
}

// ---------------------------------------------------------------------------
// Batched GEMM, all-f16 operands, f32 accumulate:
//   Y[b][m][n] = sum_k W[m][k] * X[n][k] + bias[m]     (X transposed!)
// W f16 [M][K]; X1t/X2t f16 [b][n][C] with K split at ksplit (concat inputs).
// REQUIRES K % 32 == 0, ksplit % 32 == 0.
// Wave tile 32(M) x 64(N): 12 b128 loads + 8 WMMA per K-step.
// Outputs (any may be null):
//  Yf  f32 std [b][m][n]
//  YhT f16 transposed [b][n][M]              (v8h stores)
//  YhQK f16 per-head transposed [b][h][n][64] (channel m -> h=m&3, d=m>>2)
//  YhV  f16 head-major [b][h][d][n]
// Grid: (NN/64, M/32, BB). Block: 32.
// ---------------------------------------------------------------------------
__global__ __launch_bounds__(32) void wmma_gemm_kernel(
    const _Float16* __restrict__ W, const float* __restrict__ bias,
    const _Float16* __restrict__ X1t, const _Float16* __restrict__ X2t,
    const float* __restrict__ addTo, float* __restrict__ Yf,
    _Float16* __restrict__ YhT, _Float16* __restrict__ YhQK,
    _Float16* __restrict__ YhV, int M, int K, int ksplit, int C1, int C2) {
  const int lane = threadIdx.x;
  const int hi = lane >> 4;
  const int lo = lane & 15;
  const int n0 = blockIdx.x * 64;
  const int m0 = blockIdx.y * 32;
  const int b = blockIdx.z;
  const long sY = (long)M * NN;

  v8f acc[2][4] = {};
  for (int k0 = 0; k0 < K; k0 += 32) {
    // uniform per K-step: select source block (ksplit multiple of 32)
    const _Float16* __restrict__ Xb;
    int Cs, kloc;
    if (X2t != nullptr && k0 >= ksplit) {
      Xb = X2t; Cs = C2; kloc = k0 - ksplit;
    } else {
      Xb = X1t; Cs = C1; kloc = k0;
    }
    if (k0 + 32 < K)  // speculative prefetch (gfx1250 global_prefetch_b8)
      __builtin_prefetch(Xb + ((long)b * NN + n0 + lane) * Cs + kloc + 32, 0, 1);

    v16h af[2], bf[4];
#pragma unroll
    for (int mi = 0; mi < 2; ++mi)
      af[mi] = ld_a_frag(W + (long)(m0 + 16 * mi + lo) * K + k0, hi);
#pragma unroll
    for (int ni = 0; ni < 4; ++ni)
      bf[ni] = ld_b_frag(Xb + ((long)b * NN + n0 + 16 * ni + lo) * Cs + kloc,
                         hi);
#pragma unroll
    for (int mi = 0; mi < 2; ++mi)
#pragma unroll
      for (int ni = 0; ni < 4; ++ni)
        acc[mi][ni] = wmma16(af[mi], bf[ni], acc[mi][ni]);
  }

#pragma unroll
  for (int mi = 0; mi < 2; ++mi)
#pragma unroll
    for (int ni = 0; ni < 4; ++ni) {
      const int n = n0 + 16 * ni + lo;
      const int mbase = m0 + 16 * mi + 8 * hi;
      float vv[8];
#pragma unroll
      for (int r = 0; r < 8; ++r) {
        int m = mbase + r;
        float v = acc[mi][ni][r];
        if (bias) v += bias[m];
        if (addTo) v += addTo[(long)b * sY + (long)m * NN + n];
        vv[r] = v;
      }
      if (Yf) {
#pragma unroll
        for (int r = 0; r < 8; ++r)
          Yf[(long)b * sY + (long)(mbase + r) * NN + n] = vv[r];
      }
      if (YhT) {
        v8h hv;
#pragma unroll
        for (int r = 0; r < 8; ++r) hv[r] = (_Float16)vv[r];
        *(v8h*)(YhT + ((long)b * NN + n) * M + mbase) = hv;
      }
      if (YhQK) {
#pragma unroll
        for (int r = 0; r < 8; ++r) {
          int m = mbase + r;
          int hh = m & (HH - 1), d = m >> 2;
          YhQK[((long)(b * HH + hh) * NN + n) * DIMH + d] = (_Float16)vv[r];
        }
      }
      if (YhV) {
#pragma unroll
        for (int r = 0; r < 8; ++r) {
          int m = mbase + r;
          int hh = m & (HH - 1), d = m >> 2;
          YhV[((long)(b * HH + hh) * DIMH + d) * NN + n] = (_Float16)vv[r];
        }
      }
    }
}

// ---------------------------------------------------------------------------
// Flash attention, one wave per (b, h, 16-query tile).
// qt/kt f16 [b][h][n][64]; vh f16 [b][h][d][n].
// sc[n,m] = sum_d q[n,d] k[m,d]; p = softmax_m(sc)/8; msg[d,n] = sum_m p v[d,m]
// All fragments are b128 loads. P staged via LDS (2x ds b128 reads).
// msg written f16 [b][n][256] in PERMUTED channel order c' = h*64+d
// (merge weights are column-permuted to match).
// ---------------------------------------------------------------------------
__global__ __launch_bounds__(32) void attention_kernel(
    const _Float16* __restrict__ qt, const _Float16* __restrict__ kt,
    const _Float16* __restrict__ vh, _Float16* __restrict__ msgt) {
  const int lane = threadIdx.x;
  const int hi = lane >> 4;
  const int lo = lane & 15;
  const int n0 = blockIdx.x * 16;  // query tile
  const int h = blockIdx.y;
  const int b = blockIdx.z;
  const long hnk = (long)(b * HH + h) * NN * DIMH;  // qt/kt base
  const long hdn = (long)(b * HH + h) * DIMH * NN;  // vh base

  __shared__ _Float16 pbuf[16][32];

  // Q A-fragments: row n = n0+lo, d-runs per layout (4 b128 loads total)
  const _Float16* qrow = qt + hnk + (long)(n0 + lo) * DIMH;
  v16h aq0 = ld_a_frag(qrow, hi);
  v16h aq1 = ld_a_frag(qrow + 32, hi);

  float m_run[8], l_run[8];
  v8f oacc[4] = {};
#pragma unroll
  for (int r = 0; r < 8; ++r) { m_run[r] = -1e30f; l_run[r] = 0.f; }

  for (int m0 = 0; m0 < NN; m0 += 32) {
    v8f s0 = {}, s1 = {};
#pragma unroll
    for (int kk = 0; kk < 2; ++kk) {
      const _Float16* k0r = kt + hnk + (long)(m0 + lo) * DIMH + 32 * kk;
      const _Float16* k1r = k0r + 16 * DIMH;
      v16h bk0 = ld_b_frag(k0r, hi);
      v16h bk1 = ld_b_frag(k1r, hi);
      v16h aa = (kk == 0) ? aq0 : aq1;
      s0 = wmma16(aa, bk0, s0);
      s1 = wmma16(aa, bk1, s1);
    }
#pragma unroll
    for (int r = 0; r < 8; ++r) {
      float mx = fmaxf(s0[r], s1[r]);
#pragma unroll
      for (int mask = 1; mask < 16; mask <<= 1)
        mx = fmaxf(mx, __shfl_xor(mx, mask, 32));
      float mnew = fmaxf(m_run[r], mx);
      float corr = __expf(m_run[r] - mnew);
      float p0 = __expf(s0[r] - mnew);
      float p1 = __expf(s1[r] - mnew);
      float ps = p0 + p1;
#pragma unroll
      for (int mask = 1; mask < 16; mask <<= 1) ps += __shfl_xor(ps, mask, 32);
      l_run[r] = l_run[r] * corr + ps;
      m_run[r] = mnew;
#pragma unroll
      for (int c4 = 0; c4 < 4; ++c4) oacc[c4][r] *= corr;
      pbuf[r + 8 * hi][lo] = (_Float16)p0;
      pbuf[r + 8 * hi][lo + 16] = (_Float16)p1;
    }
    __syncthreads();
    v16h ap = ld_a_frag(&pbuf[lo][0], hi);  // 2x ds b128
#pragma unroll
    for (int c4 = 0; c4 < 4; ++c4) {
      const _Float16* vrow = vh + hdn + (long)(16 * c4 + lo) * NN + m0;
      v16h bv = ld_b_frag(vrow, hi);
      oacc[c4] = wmma16(ap, bv, oacc[c4]);
    }
    __syncthreads();
  }

#pragma unroll
  for (int r = 0; r < 8; ++r) {
    int n = n0 + r + 8 * hi;
    float inv = 0.125f / l_run[r];  // post-softmax 1/sqrt(DIM)
#pragma unroll
    for (int c4 = 0; c4 < 4; ++c4) {
      int d = 16 * c4 + lo;
      // permuted channel order c' = h*64 + d  (lane-contiguous stores)
      msgt[((long)b * NN + n) * DD + h * DIMH + d] = (_Float16)(oacc[c4][r] * inv);
    }
  }
}

// ---------------------------------------------------------------------------
// BatchNorm1d (train mode): stats over (batch, length) per channel
// ---------------------------------------------------------------------------
__global__ void bn_stats_kernel(const float* __restrict__ x,
                                float* __restrict__ mean,
                                float* __restrict__ rstd, int C) {
  const int c = blockIdx.x;
  const int tid = threadIdx.x;
  const int total = BB * NN;
  float s = 0.f, s2 = 0.f;
  for (int i = tid; i < total; i += blockDim.x) {
    int b = i >> 10;
    int n = i & (NN - 1);
    float v = x[(long)b * C * NN + (long)c * NN + n];
    s += v;
    s2 += v * v;
  }
  __shared__ float sh[256], sh2[256];
  sh[tid] = s;
  sh2[tid] = s2;
  __syncthreads();
  for (int off = 128; off > 0; off >>= 1) {
    if (tid < off) {
      sh[tid] += sh[tid + off];
      sh2[tid] += sh2[tid + off];
    }
    __syncthreads();
  }
  if (tid == 0) {
    float m = sh[0] / (float)total;
    float var = sh2[0] / (float)total - m * m;
    mean[c] = m;
    rstd[c] = rsqrtf(var + 1e-5f);
  }
}

// BN + ReLU + transpose: x f32 [b][c][n] -> xt f16 [b][n][C]
// LDS-tiled 32x32 (coalesced reads and writes). Grid: (NN/32, C/32, BB).
__global__ void bn_apply_relu_t_kernel(const float* __restrict__ x,
                                       _Float16* __restrict__ xt,
                                       const float* __restrict__ mean,
                                       const float* __restrict__ rstd,
                                       const float* __restrict__ g,
                                       const float* __restrict__ beta, int C) {
  __shared__ float tile[32][33];
  const int tx = threadIdx.x, ty = threadIdx.y;
  const int n0 = blockIdx.x * 32, c0 = blockIdx.y * 32, b = blockIdx.z;
#pragma unroll
  for (int r = ty; r < 32; r += 8) {
    int c = c0 + r;
    float v = x[(long)b * C * NN + (long)c * NN + n0 + tx];
    v = (v - mean[c]) * rstd[c] * g[c] + beta[c];
    tile[r][tx] = fmaxf(v, 0.f);
  }
  __syncthreads();
#pragma unroll
  for (int r = ty; r < 32; r += 8) {
    int n = n0 + r;
    xt[((long)b * NN + n) * C + c0 + tx] = (_Float16)tile[tx][r];
  }
}

// f32 -> f16 copy (weights), grid-stride
__global__ void cvt_f16_kernel(const float* __restrict__ src,
                               _Float16* __restrict__ dst, long n) {
  long i = (long)blockIdx.x * blockDim.x + threadIdx.x;
  long stride = (long)gridDim.x * blockDim.x;
  for (; i < n; i += stride) dst[i] = (_Float16)src[i];
}

// merge weight: f16 convert + column permute c=d*4+h -> c'=h*64+d
__global__ void cvt_merge_perm_kernel(const float* __restrict__ src,
                                      _Float16* __restrict__ dst, long n) {
  long i = (long)blockIdx.x * blockDim.x + threadIdx.x;
  long stride = (long)gridDim.x * blockDim.x;
  for (; i < n; i += stride) {
    long o = i / DD;
    int c = (int)(i % DD);
    int d = c >> 2, h = c & 3;
    dst[o * DD + h * DIMH + d] = (_Float16)src[i];
  }
}

// Encoder input, f16 transposed [b][n][32] (channels 3..31 zero)
__global__ void build_enc_input_kernel(const float* __restrict__ kpts,
                                       const float* __restrict__ scores,
                                       _Float16* __restrict__ x0t) {
  int i = blockIdx.x * blockDim.x + threadIdx.x;
  if (i >= BB * NN * 32) return;
  int c = i & 31;
  int n = (i >> 5) & (NN - 1);
  int b = i >> 15;
  float v = 0.f;
  if (c == 0) {
    float kx = kpts[(long)b * NN * 2 + n * 2 + 0];
    v = (kx - 320.f) * (1.f / 448.f);  // size=[640,480]; max*0.7 = 448
  } else if (c == 1) {
    float ky = kpts[(long)b * NN * 2 + n * 2 + 1];
    v = (ky - 240.f) * (1.f / 448.f);
  } else if (c == 2) {
    v = scores[(long)b * NN + n];
  }
  x0t[i] = (_Float16)v;
}

// Zero-pad stage-0 weight (32x3) -> f16 (32x32)
__global__ void pad_w0_kernel(const float* __restrict__ w0,
                              _Float16* __restrict__ w0p) {
  int i = blockIdx.x * blockDim.x + threadIdx.x;
  if (i >= 32 * 32) return;
  int o = i >> 5, k = i & 31;
  w0p[i] = (_Float16)((k < 3) ? w0[o * 3 + k] : 0.f);
}

// ---------------------------------------------------------------------------
// Host orchestration
// ---------------------------------------------------------------------------
static inline char* carve(char*& p, size_t bytes) {
  char* r = p;
  p += (bytes + 255) & ~(size_t)255;
  return r;
}

extern "C" void kernel_launch(void* const* d_in, const int* in_sizes, int n_in,
                              void* d_out, int out_size, void* d_ws,
                              size_t ws_size, hipStream_t stream) {
  (void)in_sizes; (void)out_size; (void)ws_size;
  const float* descriptors = (const float*)d_in[0];
  const float* keypoints = (const float*)d_in[1];
  const float* scores = (const float*)d_in[2];
  // d_in[3] = image (only shape used; 480x640 fixed)

  const float *enc_w[4], *enc_b[4], *enc_g[3], *enc_be[3];
  const float *proj_w, *proj_b, *merge_w, *merge_b;
  const float *mlp1_w, *mlp1_b, *bng, *bnbeta, *mlp2_w, *mlp2_b;
  if (n_in >= 28) {  // tuples flattened into separate inputs
    for (int i = 0; i < 4; ++i) enc_w[i] = (const float*)d_in[4 + i];
    for (int i = 0; i < 4; ++i) enc_b[i] = (const float*)d_in[8 + i];
    for (int i = 0; i < 3; ++i) enc_g[i] = (const float*)d_in[12 + i];
    for (int i = 0; i < 3; ++i) enc_be[i] = (const float*)d_in[15 + i];
    proj_w = (const float*)d_in[18]; proj_b = (const float*)d_in[19];
    merge_w = (const float*)d_in[20]; merge_b = (const float*)d_in[21];
    mlp1_w = (const float*)d_in[22]; mlp1_b = (const float*)d_in[23];
    bng = (const float*)d_in[24]; bnbeta = (const float*)d_in[25];
    mlp2_w = (const float*)d_in[26]; mlp2_b = (const float*)d_in[27];
  } else {  // tuples concatenated into single buffers (18 inputs)
    const float* ew = (const float*)d_in[4];
    enc_w[0] = ew; enc_w[1] = ew + 96; enc_w[2] = ew + 96 + 2048;
    enc_w[3] = ew + 96 + 2048 + 8192;
    const float* eb = (const float*)d_in[5];
    enc_b[0] = eb; enc_b[1] = eb + 32; enc_b[2] = eb + 96; enc_b[3] = eb + 224;
    const float* eg = (const float*)d_in[6];
    enc_g[0] = eg; enc_g[1] = eg + 32; enc_g[2] = eg + 96;
    const float* ebe = (const float*)d_in[7];
    enc_be[0] = ebe; enc_be[1] = ebe + 32; enc_be[2] = ebe + 96;
    proj_w = (const float*)d_in[8]; proj_b = (const float*)d_in[9];
    merge_w = (const float*)d_in[10]; merge_b = (const float*)d_in[11];
    mlp1_w = (const float*)d_in[12]; mlp1_b = (const float*)d_in[13];
    bng = (const float*)d_in[14]; bnbeta = (const float*)d_in[15];
    mlp2_w = (const float*)d_in[16]; mlp2_b = (const float*)d_in[17];
  }

  float* descpt = (float*)d_out;  // evolving state [B,D,N] f32

  const long nProjW = (long)LL * 3 * DD * DD;
  const long nMergeW = (long)LL * DD * DD;
  const long nMlp1W = (long)LL * 2 * DD * 2 * DD;
  const long nMlp2W = (long)LL * DD * 2 * DD;

  char* wp = (char*)d_ws;
  // f16 weights
  _Float16* proj_wh = (_Float16*)carve(wp, sizeof(_Float16) * nProjW);
  _Float16* merge_whp = (_Float16*)carve(wp, sizeof(_Float16) * nMergeW);
  _Float16* mlp1_wh = (_Float16*)carve(wp, sizeof(_Float16) * nMlp1W);
  _Float16* mlp2_wh = (_Float16*)carve(wp, sizeof(_Float16) * nMlp2W);
  _Float16* enc_wh1 = (_Float16*)carve(wp, sizeof(_Float16) * 64 * 32);
  _Float16* enc_wh2 = (_Float16*)carve(wp, sizeof(_Float16) * 128 * 64);
  _Float16* enc_wh3 = (_Float16*)carve(wp, sizeof(_Float16) * 256 * 128);
  _Float16* w0h = (_Float16*)carve(wp, sizeof(_Float16) * 32 * 32);
  // activations (t = transposed [b][n][C])
  _Float16* descpt_t = (_Float16*)carve(wp, sizeof(_Float16) * BB * NN * DD);
  _Float16* xt0 = (_Float16*)carve(wp, sizeof(_Float16) * BB * NN * 256);
  _Float16* xt1 = (_Float16*)carve(wp, sizeof(_Float16) * BB * NN * 256);
  float* xf = (float*)carve(wp, sizeof(float) * BB * 256 * NN);
  _Float16* qt = (_Float16*)carve(wp, sizeof(_Float16) * BB * DD * NN);
  _Float16* kt = (_Float16*)carve(wp, sizeof(_Float16) * BB * DD * NN);
  _Float16* vh = (_Float16*)carve(wp, sizeof(_Float16) * BB * DD * NN);
  _Float16* msg_t = (_Float16*)carve(wp, sizeof(_Float16) * BB * NN * DD);
  _Float16* msg2_t = (_Float16*)carve(wp, sizeof(_Float16) * BB * NN * DD);
  float* ybuf = (float*)carve(wp, sizeof(float) * BB * 2 * DD * NN);
  _Float16* ybuf_t = (_Float16*)carve(wp, sizeof(_Float16) * BB * NN * 2 * DD);
  float* bn_mean = (float*)carve(wp, sizeof(float) * 2 * DD);
  float* bn_rstd = (float*)carve(wp, sizeof(float) * 2 * DD);

  auto cvt = [&](const float* src, _Float16* dst, long n) {
    int blocks = (int)((n + 255) / 256);
    if (blocks > 16384) blocks = 16384;
    cvt_f16_kernel<<<blocks, 256, 0, stream>>>(src, dst, n);
  };
  cvt(proj_w, proj_wh, nProjW);
  cvt(mlp1_w, mlp1_wh, nMlp1W);
  cvt(mlp2_w, mlp2_wh, nMlp2W);
  cvt(enc_w[1], enc_wh1, 64 * 32);
  cvt(enc_w[2], enc_wh2, 128 * 64);
  cvt(enc_w[3], enc_wh3, 256 * 128);
  {
    int blocks = (int)((nMergeW + 255) / 256);
    if (blocks > 16384) blocks = 16384;
    cvt_merge_perm_kernel<<<blocks, 256, 0, stream>>>(merge_w, merge_whp,
                                                      nMergeW);
  }
  pad_w0_kernel<<<4, 256, 0, stream>>>(enc_w[0], w0h);

  auto gemm = [&](const _Float16* W, const float* bias, const _Float16* X1t,
                  const _Float16* X2t, const float* addTo, float* Yf,
                  _Float16* YhT, _Float16* YhQK, _Float16* YhV, int M, int K,
                  int ksplit, int C1, int C2) {
    dim3 grid(NN / 64, M / 32, BB);
    wmma_gemm_kernel<<<grid, 32, 0, stream>>>(W, bias, X1t, X2t, addTo, Yf,
                                              YhT, YhQK, YhV, M, K, ksplit, C1,
                                              C2);
  };

  // ---------------- keypoint encoder ----------------
  build_enc_input_kernel<<<(BB * NN * 32 + 255) / 256, 256, 0, stream>>>(
      keypoints, scores, xt0);

  const int ENCC[5] = {32 /*padded from 3*/, 32, 64, 128, 256};
  const _Float16* Wl[4] = {w0h, enc_wh1, enc_wh2, enc_wh3};
  _Float16* curt = xt0;
  _Float16* nxtt = xt1;
  for (int i = 0; i < 4; ++i) {
    int Ci = ENCC[i], Co = ENCC[i + 1];
    if (i < 3) {
      gemm(Wl[i], enc_b[i], curt, nullptr, nullptr, xf, nullptr, nullptr,
           nullptr, Co, Ci, 0, Ci, 0);
      bn_stats_kernel<<<Co, 256, 0, stream>>>(xf, bn_mean, bn_rstd, Co);
      bn_apply_relu_t_kernel<<<dim3(NN / 32, Co / 32, BB), dim3(32, 8), 0,
                               stream>>>(xf, nxtt, bn_mean, bn_rstd, enc_g[i],
                                         enc_be[i], Co);
      _Float16* t = curt; curt = nxtt; nxtt = t;
    } else {
      // last conv + residual with raw descriptors -> descpt (+f16 transposed)
      gemm(Wl[i], enc_b[i], curt, nullptr, descriptors, descpt, descpt_t,
           nullptr, nullptr, Co, Ci, 0, Ci, 0);
    }
  }

  // ---------------- 18 attention layers ----------------
  for (int l = 0; l < LL; ++l) {
    const _Float16* pw = proj_wh + (long)l * 3 * DD * DD;
    const float* pb = proj_b + (long)l * 3 * DD;
    gemm(pw + 0L * DD * DD, pb + 0 * DD, descpt_t, nullptr, nullptr, nullptr,
         nullptr, qt, nullptr, DD, DD, 0, DD, 0);
    gemm(pw + 1L * DD * DD, pb + 1 * DD, descpt_t, nullptr, nullptr, nullptr,
         nullptr, kt, nullptr, DD, DD, 0, DD, 0);
    gemm(pw + 2L * DD * DD, pb + 2 * DD, descpt_t, nullptr, nullptr, nullptr,
         nullptr, nullptr, vh, DD, DD, 0, DD, 0);

    attention_kernel<<<dim3(NN / 16, HH, BB), 32, 0, stream>>>(qt, kt, vh,
                                                               msg_t);

    // merge: weights column-permuted to match msg_t channel order
    gemm(merge_whp + (long)l * DD * DD, merge_b + (long)l * DD, msg_t, nullptr,
         nullptr, nullptr, msg2_t, nullptr, nullptr, DD, DD, 0, DD, 0);

    // mlp1 on concat([descpt, msg2]) via split-K
    gemm(mlp1_wh + (long)l * 4 * DD * DD, mlp1_b + (long)l * 2 * DD, descpt_t,
         msg2_t, nullptr, ybuf, nullptr, nullptr, nullptr, 2 * DD, 2 * DD, DD,
         DD, DD);
    bn_stats_kernel<<<2 * DD, 256, 0, stream>>>(ybuf, bn_mean, bn_rstd, 2 * DD);
    bn_apply_relu_t_kernel<<<dim3(NN / 32, (2 * DD) / 32, BB), dim3(32, 8), 0,
                             stream>>>(ybuf, ybuf_t, bn_mean, bn_rstd,
                                       bng + (long)l * 2 * DD,
                                       bnbeta + (long)l * 2 * DD, 2 * DD);
    // mlp2 + residual into descpt (f32 state + f16 transposed mirror)
    gemm(mlp2_wh + (long)l * 2 * DD * DD, mlp2_b + (long)l * DD, ybuf_t,
         nullptr, descpt, descpt, descpt_t, nullptr, nullptr, DD, 2 * DD, 0,
         2 * DD, 0);
  }
}
